// LITv1Block_17557826306767
// MI455X (gfx1250) — matrix operand order repaired and checked by reading, not compiled, not run on hardware
//
#include <hip/hip_runtime.h>

typedef unsigned short u16;
typedef __attribute__((ext_vector_type(16))) __bf16 v16bf;
typedef __attribute__((ext_vector_type(8)))  float  v8f;

// ---------------------------------------------------------------- helpers
__device__ __forceinline__ u16 f2bf(float f) {
  unsigned int u = __float_as_uint(f);
  u += 0x7fffu + ((u >> 16) & 1u);          // round-to-nearest-even
  return (u16)(u >> 16);
}

union Frag { v16bf v; uint4 q[2]; };

// A-matrix 16x32 bf16 fragment: lane l holds row (l&15); K chunk
// [2*kb..2*kb+15] with kb=(l>>4)*8, contiguous in a
// row-major LDS/global row. p points at the row's K=0 element.
__device__ __forceinline__ v16bf ldfragA(const u16* p, int kb) {
  Frag f;
  f.q[0] = *(const uint4*)(p + 2 * kb);
  f.q[1] = *(const uint4*)(p + 2 * kb + 8);
  return f.v;
}
// B-matrix 32x16 bf16 fragment: lane l holds col (l&15); K=kb..kb+15
// contiguous, kb=(l>>4)*16. p points at element K=kb for this col
// (column-major staging).
__device__ __forceinline__ v16bf ldfragB(const u16* p) {
  Frag f;
  f.q[0] = *(const uint4*)(p);
  f.q[1] = *(const uint4*)(p + 8);
  return f.v;
}
__device__ __forceinline__ v8f wmma_bf16(v16bf a, v16bf b, v8f c) {
  return __builtin_amdgcn_wmma_f32_16x16x32_bf16(false, a, false, b,
                                                 (short)0, c, false, false);
}
// CDNA5 async global->LDS copy (16B/lane), tracked by ASYNCcnt.
// VDST = per-lane LDS byte address, VADDR = 64-bit global address (GV mode).
__device__ __forceinline__ void async_ld_b128(const u16* lds_p, const u16* gp) {
  unsigned lds_off = (unsigned)(size_t)lds_p;   // low 32 bits of the pointer = LDS address
  unsigned long long ga = (unsigned long long)(size_t)gp;
  asm volatile("global_load_async_to_lds_b128 %0, %1, off"
               :: "v"(lds_off), "v"(ga) : "memory");
}
__device__ __forceinline__ void wait_async0() {
  asm volatile("s_wait_asynccnt 0x0" ::: "memory");
}
// reductions across the 16 lanes of one wave32 half (xor masks < 16)
__device__ __forceinline__ float rmax16(float v) {
#pragma unroll
  for (int m = 1; m < 16; m <<= 1) v = fmaxf(v, __shfl_xor(v, m, 32));
  return v;
}
__device__ __forceinline__ float rsum16(float v) {
#pragma unroll
  for (int m = 1; m < 16; m <<= 1) v += __shfl_xor(v, m, 32);
  return v;
}

// ---------------------------------------------------------------- cast
__global__ void cast_f32_bf16(const float* __restrict__ in,
                              u16* __restrict__ out, int n) {
  int i = blockIdx.x * 256 + threadIdx.x;
  if (i < n) out[i] = f2bf(in[i]);
}

// ---------------------------------------------------------------- layernorm
// one block (256 threads) per row of 1024; emits bf16
__global__ __launch_bounds__(256)
void ln_cast(const float* __restrict__ x, const float* __restrict__ g,
             const float* __restrict__ bta, u16* __restrict__ out) {
  const int row = blockIdx.x, t = threadIdx.x;
  const float4 v = ((const float4*)(x + (size_t)row * 1024))[t];
  float s = v.x + v.y + v.z + v.w;
  float q = v.x * v.x + v.y * v.y + v.z * v.z + v.w * v.w;
#pragma unroll
  for (int m = 16; m >= 1; m >>= 1) { s += __shfl_xor(s, m, 32); q += __shfl_xor(q, m, 32); }
  __shared__ float sh[16];
  const int w = t >> 5, l = t & 31;
  if (l == 0) { sh[w] = s; sh[8 + w] = q; }
  __syncthreads();
  float ts = 0.f, tq = 0.f;
#pragma unroll
  for (int i = 0; i < 8; ++i) { ts += sh[i]; tq += sh[8 + i]; }
  const float mu = ts * (1.f / 1024.f);
  const float var = tq * (1.f / 1024.f) - mu * mu;
  const float r = rsqrtf(var + 1e-5f);
  const float4 gg = ((const float4*)g)[t];
  const float4 bb = ((const float4*)bta)[t];
  const float y0 = (v.x - mu) * r * gg.x + bb.x;
  const float y1 = (v.y - mu) * r * gg.y + bb.y;
  const float y2 = (v.z - mu) * r * gg.z + bb.z;
  const float y3 = (v.w - mu) * r * gg.w + bb.w;
  const unsigned lo = (unsigned)f2bf(y0) | ((unsigned)f2bf(y1) << 16);
  const unsigned hi = (unsigned)f2bf(y2) | ((unsigned)f2bf(y3) << 16);
  ((uint2*)(out + (size_t)row * 1024))[t] = make_uint2(lo, hi);
}

// ---------------------------------------------------------------- GEMM
// C[M,N] = A[M,K](bf16) @ B[K,N](bf16) + bias, epilogue:
//   mode 0: bf16 out     mode 1: gelu(tanh) -> bf16 out
//   mode 2: f32 out = acc + bias + resid
// 128x128 block tile, BK=32, 8 waves in a 4(M)x2(N) grid, 32x64 per wave.
// A tile staged with CDNA5 async global->LDS copies (ASYNCcnt); B tile is
// load->transpose-scatter (needs the K-major layout for B fragments).
__global__ __launch_bounds__(256)
void gemm_bf16_k(const u16* __restrict__ A, const u16* __restrict__ Bw,
                 int K, int N, const float* __restrict__ bias,
                 const float* __restrict__ resid, u16* __restrict__ outB,
                 float* __restrict__ outF, int mode) {
  __shared__ u16 sA[2][128][40];   // row-major, +8 pad
  __shared__ u16 sB[2][128][40];   // column-major [n][k], +8 pad
  const int t = threadIdx.x;
  const int m0 = blockIdx.y * 128, n0 = blockIdx.x * 128;
  const int w = t >> 5, l = t & 31, ln = l & 15, hh = l >> 4;
  const int wm = (w & 3) * 32, wn = (w >> 2) * 64;

  v8f acc[2][4];
#pragma unroll
  for (int mi = 0; mi < 2; ++mi)
#pragma unroll
    for (int nj = 0; nj < 4; ++nj)
#pragma unroll
      for (int r = 0; r < 8; ++r) acc[mi][nj][r] = 0.f;

  auto loadTiles = [&](int buf, int kt) {
    {  // A: 128 rows x 32 cols, 2 threads/row, 2x16B async per thread
      const int row = t >> 1, c0 = (t & 1) * 16;
      const u16* src = A + (size_t)(m0 + row) * K + kt * 32 + c0;
      async_ld_b128(&sA[buf][row][c0], src);
      async_ld_b128(&sA[buf][row][c0 + 8], src + 8);
    }
    {  // B: 32 rows(k) x 128 cols, coalesced read, col-major scatter
      const int k = t >> 3, c0 = (t & 7) * 16;
      const u16* src = Bw + (size_t)(kt * 32 + k) * N + n0 + c0;
      union { uint4 q[2]; u16 s[16]; } u;
      u.q[0] = ((const uint4*)src)[0];
      u.q[1] = ((const uint4*)src)[1];
#pragma unroll
      for (int i = 0; i < 16; ++i) sB[buf][c0 + i][k] = u.s[i];
    }
  };

  const int KT = K >> 5;
  loadTiles(0, 0);
  for (int kt = 0; kt < KT; ++kt) {
    wait_async0();                      // publish async A-tile writes
    __syncthreads();
    if (kt + 1 < KT) loadTiles((kt + 1) & 1, kt + 1);
    const int buf = kt & 1;
    v16bf af[2], bfv[4];
#pragma unroll
    for (int mi = 0; mi < 2; ++mi)
      af[mi] = ldfragA(&sA[buf][wm + mi * 16 + ln][0], hh * 8);
#pragma unroll
    for (int nj = 0; nj < 4; ++nj)
      bfv[nj] = ldfragB(&sB[buf][wn + nj * 16 + ln][hh * 16]);
#pragma unroll
    for (int mi = 0; mi < 2; ++mi)
#pragma unroll
      for (int nj = 0; nj < 4; ++nj)
        acc[mi][nj] = wmma_bf16(af[mi], bfv[nj], acc[mi][nj]);
  }

#pragma unroll
  for (int mi = 0; mi < 2; ++mi)
#pragma unroll
    for (int nj = 0; nj < 4; ++nj) {
      const int gcol = n0 + wn + nj * 16 + ln;
      const float bb = bias[gcol];
#pragma unroll
      for (int r = 0; r < 8; ++r) {
        const int grow = m0 + wm + mi * 16 + r + 8 * hh;
        float v = acc[mi][nj][r] + bb;
        if (mode == 0) {
          outB[(size_t)grow * N + gcol] = f2bf(v);
        } else if (mode == 1) {
          float c3 = v * v * v;
          float gl = 0.5f * v * (1.f + tanhf(0.7978845608f * (v + 0.044715f * c3)));
          outB[(size_t)grow * N + gcol] = f2bf(gl);
        } else {
          outF[(size_t)grow * N + gcol] = v + resid[(size_t)grow * N + gcol];
        }
      }
    }
}

// ---------------------------------------------------------------- attention
// one block per (b,h); 8 waves x 32 query rows; online softmax over key
// blocks of 32; qkv is [B*N, 3072] bf16 (q|k|v per 64-wide head slices).
__global__ __launch_bounds__(256)
void attn_kernel(const u16* __restrict__ qkv, const int* __restrict__ ridx,
                 const float* __restrict__ rbias, u16* __restrict__ out) {
  __shared__ u16 sK[32][72];        // [token][d], row-major (+8 pad)
  __shared__ u16 sV[64][40];        // [d][token], col-major (+8 pad)
  __shared__ u16 sP[8][32][40];     // per-wave P block, row-major (+8 pad)
  const int b = blockIdx.y, h = blockIdx.x;
  const int t = threadIdx.x, w = t >> 5, l = t & 31, ln = l & 15, hh = l >> 4;
  const float scale = 0.125f;       // Dh=64 -> 1/8

  // Q fragments, resident in registers for the whole kernel
  v16bf aq[2][2];
#pragma unroll
  for (int mi = 0; mi < 2; ++mi)
#pragma unroll
    for (int kk = 0; kk < 2; ++kk) {
      const u16* p = qkv + ((size_t)(b * 256 + w * 32 + mi * 16 + ln)) * 3072 +
                     h * 64 + kk * 32;
      aq[mi][kk] = ldfragA(p, hh * 8);
    }

  v8f o[2][4];
#pragma unroll
  for (int mi = 0; mi < 2; ++mi)
#pragma unroll
    for (int dj = 0; dj < 4; ++dj)
#pragma unroll
      for (int r = 0; r < 8; ++r) o[mi][dj][r] = 0.f;
  float mr[2][8], lr[2][8];
#pragma unroll
  for (int mi = 0; mi < 2; ++mi)
#pragma unroll
    for (int r = 0; r < 8; ++r) { mr[mi][r] = -3.0e38f; lr[mi][r] = 0.f; }

  for (int jb = 0; jb < 8; ++jb) {
    __syncthreads();
    {  // stage K block (async row-major) and V block (transposed) into LDS
      const int tok = t >> 3, c0 = (t & 7) * 8;
      const u16* srck = qkv + ((size_t)(b * 256 + jb * 32 + tok)) * 3072 +
                        1024 + h * 64 + c0;
      async_ld_b128(&sK[tok][c0], srck);
      const u16* srcv = qkv + ((size_t)(b * 256 + jb * 32 + tok)) * 3072 +
                        2048 + h * 64 + c0;
      union { uint4 q; u16 s[8]; } u;
      u.q = *(const uint4*)srcv;
#pragma unroll
      for (int i = 0; i < 8; ++i) sV[c0 + i][tok] = u.s[i];
    }
    wait_async0();                  // publish async K-tile writes
    __syncthreads();

    // S = Q @ K^T for this wave's 32 rows x 32 keys
    v16bf bk[2][2];
#pragma unroll
    for (int nj = 0; nj < 2; ++nj)
#pragma unroll
      for (int kk = 0; kk < 2; ++kk)
        bk[nj][kk] = ldfragB(&sK[nj * 16 + ln][kk * 32 + hh * 16]);
    v8f s[2][2];
#pragma unroll
    for (int mi = 0; mi < 2; ++mi)
#pragma unroll
      for (int nj = 0; nj < 2; ++nj) {
        v8f z;
#pragma unroll
        for (int r = 0; r < 8; ++r) z[r] = 0.f;
        z = wmma_bf16(aq[mi][0], bk[nj][0], z);
        s[mi][nj] = wmma_bf16(aq[mi][1], bk[nj][1], z);
      }

    // scale + gathered relative-position bias
#pragma unroll
    for (int mi = 0; mi < 2; ++mi)
#pragma unroll
      for (int r = 0; r < 8; ++r) {
        const int qrow = w * 32 + mi * 16 + r + 8 * hh;
        const int* irow = ridx + qrow * 256 + jb * 32 + ln;
#pragma unroll
        for (int nj = 0; nj < 2; ++nj) {
          const int id = irow[nj * 16];
          s[mi][nj][r] = s[mi][nj][r] * scale + rbias[id * 16 + h];
        }
      }

    // online softmax update (row stats shared by the 16 lanes of a half)
    float al[2][8];
#pragma unroll
    for (int mi = 0; mi < 2; ++mi)
#pragma unroll
      for (int r = 0; r < 8; ++r) {
        float v0 = s[mi][0][r], v1 = s[mi][1][r];
        float mx = rmax16(fmaxf(v0, v1));
        float mn = fmaxf(mr[mi][r], mx);
        float a = __expf(mr[mi][r] - mn);
        float p0 = __expf(v0 - mn), p1 = __expf(v1 - mn);
        s[mi][0][r] = p0; s[mi][1][r] = p1;
        lr[mi][r] = lr[mi][r] * a + rsum16(p0 + p1);
        mr[mi][r] = mn; al[mi][r] = a;
      }
#pragma unroll
    for (int mi = 0; mi < 2; ++mi)
#pragma unroll
      for (int dj = 0; dj < 4; ++dj)
#pragma unroll
        for (int r = 0; r < 8; ++r) o[mi][dj][r] *= al[mi][r];

    // C-layout -> A-layout through per-wave LDS
#pragma unroll
    for (int mi = 0; mi < 2; ++mi)
#pragma unroll
      for (int nj = 0; nj < 2; ++nj)
#pragma unroll
        for (int r = 0; r < 8; ++r)
          sP[w][mi * 16 + r + 8 * hh][nj * 16 + ln] = f2bf(s[mi][nj][r]);
    __syncthreads();

    // O += P @ V
    v16bf ap[2], bv[4];
#pragma unroll
    for (int mi = 0; mi < 2; ++mi)
      ap[mi] = ldfragA(&sP[w][mi * 16 + ln][0], hh * 8);
#pragma unroll
    for (int dj = 0; dj < 4; ++dj)
      bv[dj] = ldfragB(&sV[dj * 16 + ln][hh * 16]);
#pragma unroll
    for (int mi = 0; mi < 2; ++mi)
#pragma unroll
      for (int dj = 0; dj < 4; ++dj)
        o[mi][dj] = wmma_bf16(ap[mi], bv[dj], o[mi][dj]);
  }

  // normalize + merged-heads store: out[b*256+n][h*64+d] bf16
#pragma unroll
  for (int mi = 0; mi < 2; ++mi)
#pragma unroll
    for (int r = 0; r < 8; ++r) {
      const int qrow = w * 32 + mi * 16 + r + 8 * hh;
      const float inv = 1.f / lr[mi][r];
      const size_t base = ((size_t)(b * 256 + qrow)) * 1024 + h * 64 + ln;
#pragma unroll
      for (int dj = 0; dj < 4; ++dj)
        out[base + dj * 16] = f2bf(o[mi][dj][r] * inv);
    }
}

// ---------------------------------------------------------------- launch
extern "C" void kernel_launch(void* const* d_in, const int* in_sizes, int n_in,
                              void* d_out, int out_size, void* d_ws, size_t ws_size,
                              hipStream_t stream) {
  const float* x      = (const float*)d_in[0];
  const int*   ridx   = (const int*)  d_in[1];
  const float* qkv_w  = (const float*)d_in[2];
  const float* qkv_b  = (const float*)d_in[3];
  const float* proj_w = (const float*)d_in[4];
  const float* proj_b = (const float*)d_in[5];
  const float* rbias  = (const float*)d_in[6];
  const float* ln1g   = (const float*)d_in[7];
  const float* ln1b   = (const float*)d_in[8];
  const float* ln2g   = (const float*)d_in[9];
  const float* ln2b   = (const float*)d_in[10];
  const float* fc1w   = (const float*)d_in[11];
  const float* fc1b   = (const float*)d_in[12];
  const float* fc2w   = (const float*)d_in[13];
  const float* fc2b   = (const float*)d_in[14];

  char* ws = (char*)d_ws;
  // workspace layout (bytes)
  u16*   Wqkv  = (u16*)(ws + 0);                    //  6,291,456
  u16*   Wproj = (u16*)(ws + 6291456);              //  2,097,152
  u16*   Wfc1  = (u16*)(ws + 8388608);              //  8,388,608
  u16*   Wfc2  = (u16*)(ws + 16777216);             //  8,388,608
  u16*   H1    = (u16*)(ws + 25165824);             // 32 MB (reused as H2)
  u16*   QKV   = (u16*)(ws + 58720256);             // 96 MB
  u16*   ATT   = (u16*)(ws + 159383552);            // 32 MB
  float* Y1    = (float*)(ws + 192937984);          // 64 MB
  u16*   FF1   = (u16*)(ws + 260046848);            // 128 MB

  // weight casts f32 -> bf16
  cast_f32_bf16<<<(3145728 + 255) / 256, 256, 0, stream>>>(qkv_w, Wqkv, 3145728);
  cast_f32_bf16<<<(1048576 + 255) / 256, 256, 0, stream>>>(proj_w, Wproj, 1048576);
  cast_f32_bf16<<<(4194304 + 255) / 256, 256, 0, stream>>>(fc1w, Wfc1, 4194304);
  cast_f32_bf16<<<(4194304 + 255) / 256, 256, 0, stream>>>(fc2w, Wfc2, 4194304);

  // LN1 -> H1 (bf16)
  ln_cast<<<16384, 256, 0, stream>>>(x, ln1g, ln1b, H1);
  // QKV = H1 @ Wqkv + b  (bf16 out)
  gemm_bf16_k<<<dim3(24, 128), 256, 0, stream>>>(H1, Wqkv, 1024, 3072, qkv_b,
                                                 nullptr, QKV, nullptr, 0);
  // fused windowed attention with rel-pos bias -> ATT (bf16, merged heads)
  attn_kernel<<<dim3(16, 64), 256, 0, stream>>>(QKV, ridx, rbias, ATT);
  // Y1 = x + ATT @ Wproj + b  (f32 out)
  gemm_bf16_k<<<dim3(8, 128), 256, 0, stream>>>(ATT, Wproj, 1024, 1024, proj_b,
                                                x, nullptr, Y1, 2);
  // LN2 -> H1 (reuse, bf16)
  ln_cast<<<16384, 256, 0, stream>>>(Y1, ln2g, ln2b, H1);
  // FF1 = gelu(H1 @ Wfc1 + b)  (bf16 out)
  gemm_bf16_k<<<dim3(32, 128), 256, 0, stream>>>(H1, Wfc1, 1024, 4096, fc1b,
                                                 nullptr, FF1, nullptr, 1);
  // out = Y1 + FF1 @ Wfc2 + b  (f32 out)
  gemm_bf16_k<<<dim3(8, 128), 256, 0, stream>>>(FF1, Wfc2, 4096, 1024, fc2b,
                                                Y1, nullptr, (float*)d_out, 2);
}
